// EnhancedAutoformer_34359738368031
// MI455X (gfx1250) — matrix operand order repaired and checked by reading, not compiled, not run on hardware
//
#include <hip/hip_runtime.h>
#include <hip/hip_bf16.h>

// ---------------------------------------------------------------------------
// EnhancedAutoformer decoder layer for MI455X (gfx1250, wave32, WMMA).
// GEMMs: v_wmma_f32_16x16x32_bf16, bf16 operands resident in memory (half the
// HBM traffic of fp32 staging), bf16 outputs where the consumer is another
// GEMM (c_bf16 epilogue). Full interior tiles are staged HBM->LDS with
// global_load_async_to_lds_b128 (ASYNCcnt), drained after the WMMA block.
// Small ops (softmax, decomp MLP, depthwise conv) stay fp32 VALU.
// ---------------------------------------------------------------------------

typedef __attribute__((ext_vector_type(16))) __bf16        v16bf;
typedef __attribute__((ext_vector_type(8)))  float         v8f;
typedef __attribute__((ext_vector_type(4)))  float         f32x4;
typedef __attribute__((ext_vector_type(4)))  unsigned int  u32x4;

union FragAB { v16bf v; u32x4 u[2]; };
union Pack8  { u32x4 u; __bf16 h[8]; };

#define BM  128
#define BN  128
#define BK  32
#define LDK 40            // padded LDS row length (bf16 elems): 80B rows, 16B aligned

struct GemmParams {
  const void* A; long long lda, sAo, sAi;   // bf16 or fp32 (a_bf16), leading dim in elems
  const void* B; long long ldb, sBo, sBi;   // always bf16
  void*       C; long long ldc, sCo, sCi;   // fp32 or bf16 (c_bf16)
  const float* R; long long ldr, sRo, sRi;  // optional fp32 residual
  const float* alpha_ptr;                   // optional device scalar
  float alpha_const;
  int M, N, K;
  int a_bf16;     // 1: A is bf16 [M,K]; 0: A is fp32 [M,K]
  int b_kmajor;   // 0: B is bf16 [N,K] (C=A*B^T). 1: B is bf16 [K,N] (C=A*B)
  int c_bf16;     // 1: write C as bf16
  int act;        // 0 none, 1 relu
  int inner;      // inner batch count; blockIdx.z = outer*inner + inner_idx
};

__global__ __launch_bounds__(256) void gemm_wmma(GemmParams p) {
  __shared__ __bf16 As[2][BM * LDK];
  __shared__ __bf16 Bs[2][BN * LDK];

  const int zo = blockIdx.z / p.inner;
  const int zi = blockIdx.z % p.inner;
  const __bf16* Abf = (const __bf16*)p.A + (p.a_bf16 ? ((long long)zo * p.sAo + (long long)zi * p.sAi) : 0);
  const float*  Af  = (const float*)p.A  + (p.a_bf16 ? 0 : ((long long)zo * p.sAo + (long long)zi * p.sAi));
  const __bf16* B   = (const __bf16*)p.B + (long long)zo * p.sBo + (long long)zi * p.sBi;
  const long long cOff = (long long)zo * p.sCo + (long long)zi * p.sCi;
  const float* R = p.R ? (p.R + (long long)zo * p.sRo + (long long)zi * p.sRi) : nullptr;

  const int m0 = blockIdx.y * BM;
  const int n0 = blockIdx.x * BN;
  const int tid  = threadIdx.x;
  const int wave = tid >> 5;
  const int lane = tid & 31;
  const int wm = wave >> 2;          // 0..1  (64-row wave tile)
  const int wn = wave & 3;           // 0..3  (32-col wave tile)
  const int half = lane >> 4;        // 0/1
  const int l16  = lane & 15;

  const bool m_full = (m0 + BM <= p.M);
  const bool n_full = (n0 + BN <= p.N);

  // fp32-A staging coords
  const int tk = (tid & 7) * 4;      // 0..28 step 4
  const int tr = tid >> 3;           // 0..31

  // stage one K-tile into LDS buffer `buf`; returns true if async ops issued
  auto stage = [&](int buf, int k0) -> bool {
    __bf16* as = As[buf];
    __bf16* bs = Bs[buf];
    bool usedAsync = false;
    const bool k_full = (k0 + BK <= p.K);

    // ---- A ----
    if (p.a_bf16) {
      if (m_full && k_full) {
#pragma unroll
        for (int q = 0; q < 2; ++q) {
          const int c = tid + q * 256, row = c >> 2, k8 = (c & 3) * 8;
          unsigned lds = (unsigned)(size_t)&as[row * LDK + k8];
          const __bf16* gp = Abf + (long long)(m0 + row) * p.lda + k0 + k8;
          asm volatile("global_load_async_to_lds_b128 %0, %1, off"
                       :: "v"(lds), "v"(gp) : "memory");
        }
        usedAsync = true;
      } else {
#pragma unroll
        for (int q = 0; q < 2; ++q) {
          const int c = tid + q * 256, row = c >> 2, k8 = (c & 3) * 8;
          u32x4 v = {0u, 0u, 0u, 0u};
          if ((m0 + row) < p.M && (k0 + k8) < p.K)
            v = *(const u32x4*)(Abf + (long long)(m0 + row) * p.lda + k0 + k8);
          *(u32x4*)&as[row * LDK + k8] = v;
        }
      }
    } else {
#pragma unroll
      for (int rr = 0; rr < 4; ++rr) {
        const int row = tr + rr * 32;
        f32x4 val = {0.f, 0.f, 0.f, 0.f};
        if ((m0 + row) < p.M && (k0 + tk) < p.K)
          val = *(const f32x4*)(Af + (long long)(m0 + row) * p.lda + k0 + tk);
        __bf16* dst = &as[row * LDK + tk];
        dst[0] = (__bf16)val.x; dst[1] = (__bf16)val.y;
        dst[2] = (__bf16)val.z; dst[3] = (__bf16)val.w;
      }
    }

    // ---- B ----
    if (!p.b_kmajor) {
      if (n_full && k_full) {
#pragma unroll
        for (int q = 0; q < 2; ++q) {
          const int c = tid + q * 256, row = c >> 2, k8 = (c & 3) * 8;
          unsigned lds = (unsigned)(size_t)&bs[row * LDK + k8];
          const __bf16* gp = B + (long long)(n0 + row) * p.ldb + k0 + k8;
          asm volatile("global_load_async_to_lds_b128 %0, %1, off"
                       :: "v"(lds), "v"(gp) : "memory");
        }
        usedAsync = true;
      } else {
#pragma unroll
        for (int q = 0; q < 2; ++q) {
          const int c = tid + q * 256, row = c >> 2, k8 = (c & 3) * 8;
          u32x4 v = {0u, 0u, 0u, 0u};
          if ((n0 + row) < p.N && (k0 + k8) < p.K)
            v = *(const u32x4*)(B + (long long)(n0 + row) * p.ldb + k0 + k8);
          *(u32x4*)&bs[row * LDK + k8] = v;
        }
      }
    } else {
      // B[k][n] -> Bs[n][k] transpose staging (8 bf16 along n per chunk)
#pragma unroll
      for (int q = 0; q < 2; ++q) {
        const int c = tid + q * 256, kk = c >> 4, n8 = (c & 15) * 8;
        Pack8 v; v.u = (u32x4){0u, 0u, 0u, 0u};
        if ((k0 + kk) < p.K && (n0 + n8) < p.N)
          v.u = *(const u32x4*)(B + (long long)(k0 + kk) * p.ldb + n0 + n8);
#pragma unroll
        for (int i = 0; i < 8; ++i) bs[(n8 + i) * LDK + kk] = v.h[i];
      }
    }
    return usedAsync;
  };

  v8f acc[4][2] = {};
  const int nk = (p.K + BK - 1) / BK;

  if (stage(0, 0))
    asm volatile("s_wait_asynccnt 0x0" ::: "memory");
  __syncthreads();

  for (int t = 0; t < nk; ++t) {
    const int cur = t & 1;
    bool pend = false;
    if (t + 1 < nk) pend = stage(cur ^ 1, (t + 1) * BK);  // issue next-tile staging

    // ---- fragment loads per CDNA5 16-bit WMMA layouts ----
    FragAB a[4], b[2];
#pragma unroll
    for (int i = 0; i < 4; ++i) {
      const __bf16* ar = &As[cur][(wm * 64 + i * 16 + l16) * LDK];
      a[i].u[0] = *(const u32x4*)(ar + half * 8);        // K = half*8 + 0..7
      a[i].u[1] = *(const u32x4*)(ar + 16 + half * 8);   // K = 16 + half*8 + 0..7
    }
#pragma unroll
    for (int j = 0; j < 2; ++j) {
      const __bf16* br = &Bs[cur][(wn * 32 + j * 16 + l16) * LDK];
      b[j].u[0] = *(const u32x4*)(br + half * 8);        // K = half*8 + 0..7
      b[j].u[1] = *(const u32x4*)(br + 16 + half * 8);   // K = 16 + half*8 + 0..7
    }
#pragma unroll
    for (int i = 0; i < 4; ++i)
#pragma unroll
      for (int j = 0; j < 2; ++j)
        acc[i][j] = __builtin_amdgcn_wmma_f32_16x16x32_bf16(
            false, a[i].v, false, b[j].v, (short)0, acc[i][j], false, false);

    if (pend) asm volatile("s_wait_asynccnt 0x0" ::: "memory");
    __syncthreads();
  }

  // ---- epilogue: C = act(res + alpha * acc) ----
  const float alpha = p.alpha_const * (p.alpha_ptr ? p.alpha_ptr[0] : 1.0f);
#pragma unroll
  for (int i = 0; i < 4; ++i)
#pragma unroll
    for (int j = 0; j < 2; ++j)
#pragma unroll
      for (int r = 0; r < 8; ++r) {
        const int gm = m0 + wm * 64 + i * 16 + r + 8 * half;
        const int gn = n0 + wn * 32 + j * 16 + l16;
        if (gm < p.M && gn < p.N) {
          float vv = alpha * acc[i][j][r];
          if (R) vv += R[(long long)gm * p.ldr + gn];
          if (p.act == 1) vv = fmaxf(vv, 0.f);
          if (p.c_bf16) ((__bf16*)p.C)[cOff + (long long)gm * p.ldc + gn] = (__bf16)vv;
          else          ((float*)p.C)[cOff + (long long)gm * p.ldc + gn] = vv;
        }
      }
}

// ---------------------------------------------------------------------------
// fp32 -> bf16 conversion (n must be a multiple of 8)
// ---------------------------------------------------------------------------
__global__ void f32_to_bf16(const float* __restrict__ s, __bf16* __restrict__ d, long long n) {
  const long long i = ((long long)blockIdx.x * blockDim.x + threadIdx.x) * 8;
  if (i + 8 <= n) {
    f32x4 a = *(const f32x4*)(s + i);
    f32x4 b = *(const f32x4*)(s + i + 4);
    Pack8 o;
#pragma unroll
    for (int j = 0; j < 4; ++j) { o.h[j] = (__bf16)a[j]; o.h[4 + j] = (__bf16)b[j]; }
    *(u32x4*)(d + i) = o.u;
  }
}

// ---------------------------------------------------------------------------
// Row softmax: one wave per row, rows contiguous with leading dim n.
// ---------------------------------------------------------------------------
__global__ __launch_bounds__(256) void softmax_rows(float* S, long long rows, int n) {
  const int wave = threadIdx.x >> 5, lane = threadIdx.x & 31;
  const long long row = (long long)blockIdx.x * 8 + wave;
  if (row >= rows) return;
  float* s = S + row * (long long)n;
  float buf[16];
  float m = -1e30f;
  int cnt = 0;
  for (int j = lane; j < n; j += 32) { float v = s[j]; buf[cnt++] = v; m = fmaxf(m, v); }
  for (int o = 16; o > 0; o >>= 1) m = fmaxf(m, __shfl_xor(m, o, 32));
  float sum = 0.f; cnt = 0;
  for (int j = lane; j < n; j += 32) { float e = expf(buf[cnt] - m); buf[cnt] = e; sum += e; ++cnt; }
  for (int o = 16; o > 0; o >>= 1) sum += __shfl_xor(sum, o, 32);
  const float inv = 1.f / sum;
  cnt = 0;
  for (int j = lane; j < n; j += 32) s[j] = buf[cnt++] * inv;
}

// ---------------------------------------------------------------------------
// Decomposition helpers
// ---------------------------------------------------------------------------
__global__ void mean_l(const float* x, float* xg, int L, int Dd) {
  const int d = blockIdx.x * blockDim.x + threadIdx.x;
  const int b = blockIdx.y;
  if (d < Dd) {
    float s = 0.f;
    const float* p = x + (long long)b * L * Dd + d;
    for (int l = 0; l < L; ++l) s += p[(long long)l * Dd];
    xg[b * Dd + d] = s / (float)L;
  }
}

__global__ __launch_bounds__(256) void decomp_mlp(const float* xg, const float* w1,
                                                  const float* b1, const float* w2,
                                                  const float* b2, int* kk, int L) {
  const int b = blockIdx.x, t = threadIdx.x;   // 256 threads == hidden dim (D/2)
  __shared__ float sx[512];
  __shared__ float red[256];
  for (int i = t; i < 512; i += 256) sx[i] = xg[b * 512 + i];
  __syncthreads();
  float h = 0.f;
  const float* w = w1 + t * 512;
  for (int i = 0; i < 512; ++i) h += sx[i] * w[i];
  h = fmaxf(h + b1[t], 0.f);
  red[t] = h * w2[t];
  __syncthreads();
  for (int s = 128; s > 0; s >>= 1) { if (t < s) red[t] += red[t + s]; __syncthreads(); }
  if (t == 0) {
    const float logit = 1.f / (1.f + expf(-(red[0] + b2[0])));
    const float kf = logit * 45.f + 5.f;           // logit*(MAXK-5)+5
    int k = (int)rintf(kf);
    const int kmax = min(50, L / 2);
    k = max(3, min(k, kmax));
    if ((k & 1) == 0) k -= 1;
    k = max(k, 3);
    kk[b] = k;
  }
}

// Per (b,d): masked softmax over first k taps of tw[d,:], recentered into 51-wide window.
__global__ void build_wf(const float* tw, const int* kk, float* wf) {
  const int idx = blockIdx.x * blockDim.x + threadIdx.x;  // b*512 + d
  if (idx >= 32 * 512) return;
  const int b = idx >> 9, d = idx & 511;
  const int k = kk[b];
  const float* t = tw + d * 50;
  float m = -1e30f;
  for (int i = 0; i < k; ++i) m = fmaxf(m, t[i]);
  float s = 0.f;
  for (int i = 0; i < k; ++i) s += expf(t[i] - m);
  const float inv = 1.f / s;
  const int off = 25 - k / 2;
  float* o = wf + (long long)idx * 51;
  for (int j = 0; j < 51; ++j) {
    const int i = j - off;
    o[j] = (i >= 0 && i < k) ? expf(t[i] - m) * inv : 0.f;
  }
}

// Depthwise conv with replicate padding: season = x - trend, trend written or accumulated.
__global__ __launch_bounds__(256) void dwconv(const float* x, const float* wf,
                                              float* season, float* trend,
                                              int L, int addTrend) {
  __shared__ float xs[82][64];   // (32 + 50) x 64 = ~21 KB
  const int b  = blockIdx.z;
  const int l0 = blockIdx.x * 32;
  const int d0 = blockIdx.y * 64;
  const int t  = threadIdx.x;
  for (int i = t; i < 82 * 64; i += 256) {
    const int r = i >> 6, c = i & 63;
    int l = l0 + r - 25;
    l = min(max(l, 0), L - 1);     // edge (replicate) padding
    xs[r][c] = x[((long long)b * L + l) * 512 + d0 + c];
  }
  __syncthreads();
  const int c  = t & 63;
  const int lr = t >> 6;   // 4 l's per thread group
  const float* w = wf + ((long long)b * 512 + d0 + c) * 51;
  for (int ll = lr; ll < 32; ll += 4) {
    if (l0 + ll >= L) break;
    float s = 0.f;
#pragma unroll
    for (int j = 0; j < 51; ++j) s += w[j] * xs[ll + j][c];
    const long long oi = ((long long)b * L + l0 + ll) * 512 + d0 + c;
    season[oi] = xs[ll + 25][c] - s;
    if (addTrend) trend[oi] += s; else trend[oi] = s;
  }
}

// Circular 3-tap window gather, written directly as bf16 GEMM operand:
// wins[b,l,d*3+j] = tr[b,(l+j-1) mod L, d]
__global__ void build_wins(const float* tr, __bf16* wins, int L) {
  const long long idx = (long long)blockIdx.x * 256 + threadIdx.x;
  const long long total = (long long)32 * L * 1536;
  if (idx >= total) return;
  const int kkk = (int)(idx % 1536);
  const long long bl = idx / 1536;
  const int l = (int)(bl % L);
  const int b = (int)(bl / L);
  const int j = kkk % 3, d = kkk / 3;
  int ls = l + j - 1;
  ls = (ls + L) % L;
  wins[idx] = (__bf16)tr[((long long)b * L + ls) * 512 + d];
}

// ---------------------------------------------------------------------------
// Host side
// ---------------------------------------------------------------------------
static void gemm(hipStream_t st,
                 const void* A, long long lda, long long sAo, long long sAi,
                 const void* B, long long ldb, long long sBo, long long sBi,
                 void* C, long long ldc, long long sCo, long long sCi,
                 int M, int N, int K, int batches, int inner,
                 const float* R, long long ldr, long long sRo, long long sRi,
                 const float* alpha_ptr, float alpha_const,
                 int a_bf16, int b_kmajor, int c_bf16, int act) {
  GemmParams p;
  p.A = A; p.lda = lda; p.sAo = sAo; p.sAi = sAi;
  p.B = B; p.ldb = ldb; p.sBo = sBo; p.sBi = sBi;
  p.C = C; p.ldc = ldc; p.sCo = sCo; p.sCi = sCi;
  p.R = R; p.ldr = ldr; p.sRo = sRo; p.sRi = sRi;
  p.alpha_ptr = alpha_ptr; p.alpha_const = alpha_const;
  p.M = M; p.N = N; p.K = K;
  p.a_bf16 = a_bf16; p.b_kmajor = b_kmajor; p.c_bf16 = c_bf16; p.act = act; p.inner = inner;
  dim3 g((N + BN - 1) / BN, (M + BM - 1) / BM, batches);
  gemm_wmma<<<g, 256, 0, st>>>(p);
}

static void cvt(hipStream_t st, const float* s, __bf16* d, long long n) {
  f32_to_bf16<<<(int)((n / 8 + 255) / 256), 256, 0, st>>>(s, d, n);
}

extern "C" void kernel_launch(void* const* d_in, const int* in_sizes, int n_in,
                              void* d_out, int out_size, void* d_ws, size_t ws_size,
                              hipStream_t stream) {
  (void)in_sizes; (void)n_in; (void)out_size; (void)ws_size;
  const int Bn = 32, L = 336, Lc = 192, D = 512, H = 8, DH = 64, DFF = 2048;
  const int M1 = Bn * L;          // 10752
  const int Mc = Bn * Lc;         // 6144
  const long long SZ = (long long)M1 * D;   // 5,505,024

  // inputs (setup_inputs dict order, params depth-first)
  const float* xin   = (const float*)d_in[0];
  const float* cross = (const float*)d_in[1];
  const float* w_f32[12] = {
    (const float*)d_in[2],  (const float*)d_in[3],  (const float*)d_in[4],  (const float*)d_in[5],
    (const float*)d_in[6],  (const float*)d_in[7],  (const float*)d_in[8],  (const float*)d_in[9],
    (const float*)d_in[10], (const float*)d_in[11], (const float*)d_in[12], (const float*)d_in[13]};
  const long long w_sz[12] = {262144,262144,262144,262144, 262144,262144,262144,262144,
                              1048576, 1048576, 786432, 262144};
  const float* self_scale  = (const float*)d_in[14];
  const float* cross_scale = (const float*)d_in[15];
  const float* dec_tw[3] = {(const float*)d_in[16], (const float*)d_in[21], (const float*)d_in[26]};
  const float* dec_w1[3] = {(const float*)d_in[17], (const float*)d_in[22], (const float*)d_in[27]};
  const float* dec_b1[3] = {(const float*)d_in[18], (const float*)d_in[23], (const float*)d_in[28]};
  const float* dec_w2[3] = {(const float*)d_in[19], (const float*)d_in[24], (const float*)d_in[29]};
  const float* dec_b2[3] = {(const float*)d_in[20], (const float*)d_in[25], (const float*)d_in[30]};

  // ---- workspace layout ----
  float* ws = (float*)d_ws;
  float* xa  = ws;                       // residual stream A          [SZ]
  float* xb  = xa + SZ;                  // residual stream B          [SZ]
  float* trd = xb + SZ;                  // accumulated trend          [SZ]
  float* Sb  = trd + SZ;                 // attention scores           [28,901,376]
  float* wf  = Sb + 28901376;            // depthwise kernels          [835,584]
  float* xg  = wf + 835584;              // per-sample mean            [16,384]
  int*   kkb = (int*)(xg + 16384);       // kernel sizes               [32]
  __bf16* bh = (__bf16*)(xg + 16384 + 32);
  __bf16* qbf    = bh;                   // [SZ]
  __bf16* kbf    = qbf + SZ;             // [SZ]
  __bf16* vbf    = kbf + SZ;             // [SZ]
  __bf16* obf    = vbf + SZ;             // [SZ]
  __bf16* xinbf  = obf + SZ;             // [SZ]
  __bf16* xbbf   = xinbf + SZ;           // [SZ]
  __bf16* cbf    = xbbf + SZ;            // [SZ]
  __bf16* crossbf= cbf + SZ;             // [3,145,728]
  __bf16* hbf    = crossbf + 3145728;    // [22,020,096]
  __bf16* winsbf = hbf + 22020096;       // [16,515,072]
  __bf16* wbf[12];
  { __bf16* pW = winsbf + 16515072;
    for (int i = 0; i < 12; ++i) { wbf[i] = pW; pW += w_sz[i]; } }

  float* out_season = (float*)d_out;
  float* out_trend  = (float*)d_out + SZ;

  const long long sLD = (long long)L * D, sCD = (long long)Lc * D;

  // ---- one-time operand conversions ----
  for (int i = 0; i < 12; ++i) cvt(stream, w_f32[i], wbf[i], w_sz[i]);
  cvt(stream, xin,   xinbf,   SZ);
  cvt(stream, cross, crossbf, (long long)Mc * D);

  // ---------------- self-attention ----------------
  gemm(stream, xinbf, D,0,0, wbf[0], D,0,0, qbf, D,0,0, M1, D, D, 1,1, nullptr,0,0,0, nullptr,1.f, 1,0,1,0);
  gemm(stream, xinbf, D,0,0, wbf[1], D,0,0, kbf, D,0,0, M1, D, D, 1,1, nullptr,0,0,0, nullptr,1.f, 1,0,1,0);
  gemm(stream, xinbf, D,0,0, wbf[2], D,0,0, vbf, D,0,0, M1, D, D, 1,1, nullptr,0,0,0, nullptr,1.f, 1,0,1,0);
  {
    const long long sS_h = (long long)L * L, sS_b = (long long)H * L * L;
    gemm(stream, qbf, D, sLD, DH, kbf, D, sLD, DH, Sb, L, sS_b, sS_h,
         L, L, DH, Bn * H, H, nullptr,0,0,0, nullptr, 0.125f, 1,0,0,0);
    softmax_rows<<<(Bn * H * L + 7) / 8, 256, 0, stream>>>(Sb, (long long)Bn * H * L, L);
    gemm(stream, Sb, L, sS_b, sS_h, vbf, D, sLD, DH, obf, D, sLD, DH,
         L, DH, L, Bn * H, H, nullptr,0,0,0, nullptr, 1.f, 0,1,1,0);   // fp32 A, k-major B, bf16 C
  }
  gemm(stream, obf, D,0,0, wbf[3], D,0,0, xa, D,0,0, M1, D, D, 1,1,
       xin, D,0,0, self_scale, 1.f, 1,0,0,0);

  // ---------------- decomp 1 ----------------
  {
    dim3 gm(2, Bn); mean_l<<<gm, 256, 0, stream>>>(xa, xg, L, D);
    decomp_mlp<<<Bn, 256, 0, stream>>>(xg, dec_w1[0], dec_b1[0], dec_w2[0], dec_b2[0], kkb, L);
    build_wf<<<(Bn * D + 255) / 256, 256, 0, stream>>>(dec_tw[0], kkb, wf);
    dim3 gd((L + 31) / 32, D / 64, Bn);
    dwconv<<<gd, 256, 0, stream>>>(xa, wf, xb, trd, L, 0);
  }

  // ---------------- cross-attention ----------------
  cvt(stream, xb, xbbf, SZ);
  gemm(stream, xbbf,    D,0,0, wbf[4], D,0,0, qbf, D,0,0, M1, D, D, 1,1, nullptr,0,0,0, nullptr,1.f, 1,0,1,0);
  gemm(stream, crossbf, D,0,0, wbf[5], D,0,0, kbf, D,0,0, Mc, D, D, 1,1, nullptr,0,0,0, nullptr,1.f, 1,0,1,0);
  gemm(stream, crossbf, D,0,0, wbf[6], D,0,0, vbf, D,0,0, Mc, D, D, 1,1, nullptr,0,0,0, nullptr,1.f, 1,0,1,0);
  {
    const long long sS_h = (long long)L * Lc, sS_b = (long long)H * L * Lc;
    gemm(stream, qbf, D, sLD, DH, kbf, D, sCD, DH, Sb, Lc, sS_b, sS_h,
         L, Lc, DH, Bn * H, H, nullptr,0,0,0, nullptr, 0.125f, 1,0,0,0);
    softmax_rows<<<(Bn * H * L + 7) / 8, 256, 0, stream>>>(Sb, (long long)Bn * H * L, Lc);
    gemm(stream, Sb, Lc, sS_b, sS_h, vbf, D, sCD, DH, obf, D, sLD, DH,
         L, DH, Lc, Bn * H, H, nullptr,0,0,0, nullptr, 1.f, 0,1,1,0);
  }
  gemm(stream, obf, D,0,0, wbf[7], D,0,0, xa, D,0,0, M1, D, D, 1,1,
       xb, D,0,0, cross_scale, 1.f, 1,0,0,0);

  // ---------------- decomp 2 ----------------
  {
    dim3 gm(2, Bn); mean_l<<<gm, 256, 0, stream>>>(xa, xg, L, D);
    decomp_mlp<<<Bn, 256, 0, stream>>>(xg, dec_w1[1], dec_b1[1], dec_w2[1], dec_b2[1], kkb, L);
    build_wf<<<(Bn * D + 255) / 256, 256, 0, stream>>>(dec_tw[1], kkb, wf);
    dim3 gd((L + 31) / 32, D / 64, Bn);
    dwconv<<<gd, 256, 0, stream>>>(xa, wf, xb, trd, L, 1);
  }

  // ---------------- FFN ----------------
  cvt(stream, xb, xbbf, SZ);
  gemm(stream, xbbf, D,0,0, wbf[8], D,0,0, hbf, DFF,0,0, M1, DFF, D, 1,1,
       nullptr,0,0,0, nullptr, 1.f, 1,0,1,1);                   // relu, bf16 out
  gemm(stream, hbf, DFF,0,0, wbf[9], DFF,0,0, xa, D,0,0, M1, D, DFF, 1,1,
       xb, D,0,0, nullptr, 1.f, 1,0,0,0);                       // + residual

  // ---------------- decomp 3 (seasonal -> d_out) ----------------
  {
    dim3 gm(2, Bn); mean_l<<<gm, 256, 0, stream>>>(xa, xg, L, D);
    decomp_mlp<<<Bn, 256, 0, stream>>>(xg, dec_w1[2], dec_b1[2], dec_w2[2], dec_b2[2], kkb, L);
    build_wf<<<(Bn * D + 255) / 256, 256, 0, stream>>>(dec_tw[2], kkb, wf);
    dim3 gd((L + 31) / 32, D / 64, Bn);
    dwconv<<<gd, 256, 0, stream>>>(xa, wf, out_season, trd, L, 1);
  }

  // ---------------- trend projection ----------------
  {
    const long long total = (long long)Bn * L * 1536;
    build_wins<<<(int)((total + 255) / 256), 256, 0, stream>>>(trd, winsbf, L);
    gemm(stream, winsbf, 1536,0,0, wbf[10], 1536,0,0, cbf, D,0,0, M1, D, 1536, 1,1,
         nullptr,0,0,0, nullptr, 1.f, 1,0,1,1);                 // relu, bf16 out
    gemm(stream, cbf, D,0,0, wbf[11], D,0,0, out_trend, D,0,0, M1, D, D, 1,1,
         nullptr,0,0,0, nullptr, 1.f, 1,0,0,0);
  }
}
